// GraphAttention_82257213653581
// MI455X (gfx1250) — matrix operand order repaired and checked by reading, not compiled, not run on hardware
//
#include <hip/hip_runtime.h>
#include <hip/hip_bf16.h>

// ---------------- problem constants ----------------
constexpr int Nn   = 8192;   // nodes
constexpr int ATOM = 64;     // input features
constexpr int HIDc = 128;    // hidden features
constexpr int Gc   = 256;    // graphs
constexpr float NEG = 0.2f;  // leakyrelu slope

typedef float    v2f  __attribute__((ext_vector_type(2)));
typedef float    v8f  __attribute__((ext_vector_type(8)));
typedef _Float16 v16h __attribute__((ext_vector_type(16)));
typedef _Float16 h8   __attribute__((ext_vector_type(8)));

__device__ __forceinline__ float lrelu(float x) { return x > 0.f ? x : NEG * x; }

// ---------------------------------------------------------------
// Fold W[1024,128] -> Weff[128,128]: tile(h,8) @ W == h @ sum(blocks)
// ---------------------------------------------------------------
__global__ __launch_bounds__(128) void k_fold(const float* __restrict__ Wbig,
                                              float* __restrict__ Weff) {
    int idx = blockIdx.x * blockDim.x + threadIdx.x;   // k*128 + c
    if (idx >= HIDc * HIDc) return;
    int k = idx >> 7, c = idx & 127;
    float sum = 0.f;
#pragma unroll
    for (int b = 0; b < 8; ++b) sum += Wbig[(size_t)(b * HIDc + k) * HIDc + c];
    Weff[idx] = sum;
}

// ---------------------------------------------------------------
// Feature GEMM, exact fp32 via V_WMMA_F32_16X16X4_F32.
// One wave per 16-row block, computes all 128 output columns.
// Writes h (fp32, row-major) and ht (f16, transposed [HID][N]) for attention B.
// A 16x4 f32 layout: lane<16 holds A[M=lane][K=2*hi +0/1] in v2f.
// B 4x16  f32 layout: lane n holds B[K=2*hi+0/1][N=n%16].
// ---------------------------------------------------------------
__global__ __launch_bounds__(32) void k_gemm(const float* __restrict__ in,
                                             const float* __restrict__ W,
                                             float* __restrict__ h,
                                             _Float16* __restrict__ ht,
                                             int Kin, int relu_in) {
    const int lane = threadIdx.x & 31;
    const int lm = lane & 15, hi = lane >> 4;
    const int r0 = blockIdx.x * 16;
    const float* inr = in + (size_t)(r0 + lm) * Kin;

    v8f zero = {0.f, 0.f, 0.f, 0.f, 0.f, 0.f, 0.f, 0.f};
    v8f acc[8];
#pragma unroll
    for (int t = 0; t < 8; ++t) acc[t] = zero;

    for (int k = 0; k < Kin; k += 4) {
        float a0 = inr[k + 2 * hi + 0];
        float a1 = inr[k + 2 * hi + 1];
        if (relu_in) { a0 = fmaxf(a0, 0.f); a1 = fmaxf(a1, 0.f); }
        v2f A = {a0, a1};
        const float* w0 = W + (size_t)(k + 2 * hi) * HIDc;
        const float* w1 = w0 + HIDc;
#pragma unroll
        for (int t = 0; t < 8; ++t) {
            v2f B = {w0[t * 16 + lm], w1[t * 16 + lm]};
            acc[t] = __builtin_amdgcn_wmma_f32_16x16x4_f32(
                false, A, false, B, (short)0, acc[t], false, false);
        }
    }
    // C layout: VGPR r -> rows (r, 8+r); lanes 0-15 cols n, lanes 16-31 cols n (M=8+r)
#pragma unroll
    for (int r = 0; r < 8; ++r) {
        int row = r0 + r + 8 * hi;
#pragma unroll
        for (int t = 0; t < 8; ++t) {
            float v = acc[t][r];
            h[(size_t)row * HIDc + t * 16 + lm] = v;
            ht[(size_t)(t * 16 + lm) * Nn + row] = (_Float16)v;
        }
    }
}

// ---------------------------------------------------------------
// s = h @ a_src, d = h @ a_dst (per-row dots)
// ---------------------------------------------------------------
__global__ __launch_bounds__(256) void k_sd(const float* __restrict__ h,
                                            const float* __restrict__ asrc,
                                            const float* __restrict__ adst,
                                            float* __restrict__ s,
                                            float* __restrict__ d) {
    int i = blockIdx.x * blockDim.x + threadIdx.x;
    if (i >= Nn) return;
    const float4* hv = (const float4*)(h + (size_t)i * HIDc);
    const float4* sv = (const float4*)asrc;
    const float4* dv = (const float4*)adst;
    float ss = 0.f, dd = 0.f;
#pragma unroll 8
    for (int c = 0; c < HIDc / 4; ++c) {
        float4 hx = hv[c], sx = sv[c], dx = dv[c];
        ss += hx.x * sx.x + hx.y * sx.y + hx.z * sx.z + hx.w * sx.w;
        dd += hx.x * dx.x + hx.y * dx.y + hx.z * dx.z + hx.w * dx.w;
    }
    s[i] = ss;
    d[i] = dd;
}

// ---------------------------------------------------------------
// Global max of d (single block) -> safe softmax shift
// ---------------------------------------------------------------
__global__ __launch_bounds__(1024) void k_maxd(const float* __restrict__ d,
                                               float* __restrict__ Dmax) {
    __shared__ float red[1024];
    int t = threadIdx.x;
    float m = -1e30f;
    for (int i = t; i < Nn; i += 1024) m = fmaxf(m, d[i]);
    red[t] = m;
    __syncthreads();
    for (int sft = 512; sft > 0; sft >>= 1) {
        if (t < sft) red[t] = fmaxf(red[t], red[t + sft]);
        __syncthreads();
    }
    if (t == 0) Dmax[0] = red[0];
}

// ---------------------------------------------------------------
// Fused flash-GAT attention: one streaming pass over adj per row-block.
// c_i = lrelu(s_i + max_j d_j) >= row max (lrelu monotone) -> shift-invariant
// softmax needs no online rescaling. P tile built directly in f16 A-layout,
// B tiles are contiguous loads from transposed f16 h. out = (P @ h) / rowsum.
// adj stream is the HBM-bound path -> speculative global_prefetch_b8 ahead.
// ---------------------------------------------------------------
__global__ __launch_bounds__(256) void k_attn(const float* __restrict__ adj,
                                              const float* __restrict__ s,
                                              const float* __restrict__ d,
                                              const float* __restrict__ Dmax,
                                              const _Float16* __restrict__ ht,
                                              float* __restrict__ out) {
    const int lane = threadIdx.x & 31;
    const int lm = lane & 15, hi = lane >> 4;
    const int wave = threadIdx.x >> 5;
    const int r0 = (blockIdx.x * 8 + wave) * 16;
    const int row = r0 + lm;

    const float si = s[row];
    const float ci = lrelu(si + Dmax[0]);
    const float* arow = adj + (size_t)row * Nn;

    const int ka = hi * 8;       // A-matrix low-K offset  (lanes0-15: 0..7 ; lanes16-31: 8..15)
    const int kb = 16 + hi * 8;  // A-matrix high-K offset (16..23 / 24..31)
    const int bk = hi * 8;       // B-matrix K offset      (lanes0-15: K 0..7,16..23 ; 8..15,24..31)

    v8f zero = {0.f, 0.f, 0.f, 0.f, 0.f, 0.f, 0.f, 0.f};
    v8f acc[8];
#pragma unroll
    for (int t = 0; t < 8; ++t) acc[t] = zero;
    float lsum = 0.f;

    const _Float16* htb[8];
#pragma unroll
    for (int t = 0; t < 8; ++t) htb[t] = ht + (size_t)(t * 16 + lm) * Nn;

    for (int j = 0; j < Nn; j += 32) {
        // prefetch the adj stream 8 iterations (1 KB/row) ahead; speculative,
        // silently dropped past the end of the buffer
        __builtin_prefetch(arow + j + 256, 0, 1);

        float4 a0 = *(const float4*)(arow + j + ka);
        float4 a1 = *(const float4*)(arow + j + ka + 4);
        float4 a2 = *(const float4*)(arow + j + kb);
        float4 a3 = *(const float4*)(arow + j + kb + 4);
        float4 d0 = *(const float4*)(d + j + ka);
        float4 d1 = *(const float4*)(d + j + ka + 4);
        float4 d2 = *(const float4*)(d + j + kb);
        float4 d3 = *(const float4*)(d + j + kb + 4);

        auto pe = [&](float av, float dv) -> _Float16 {
            float p = (av > 0.f) ? __expf(lrelu(si + dv) - ci) : 0.f;
            lsum += p;
            return (_Float16)p;
        };
        v16h A;
        A[0]  = pe(a0.x, d0.x); A[1]  = pe(a0.y, d0.y);
        A[2]  = pe(a0.z, d0.z); A[3]  = pe(a0.w, d0.w);
        A[4]  = pe(a1.x, d1.x); A[5]  = pe(a1.y, d1.y);
        A[6]  = pe(a1.z, d1.z); A[7]  = pe(a1.w, d1.w);
        A[8]  = pe(a2.x, d2.x); A[9]  = pe(a2.y, d2.y);
        A[10] = pe(a2.z, d2.z); A[11] = pe(a2.w, d2.w);
        A[12] = pe(a3.x, d3.x); A[13] = pe(a3.y, d3.y);
        A[14] = pe(a3.z, d3.z); A[15] = pe(a3.w, d3.w);

#pragma unroll
        for (int t = 0; t < 8; ++t) {
            h8 b0 = *(const h8*)(htb[t] + j + bk);
            h8 b1 = *(const h8*)(htb[t] + j + bk + 16);
            v16h B;
#pragma unroll
            for (int e = 0; e < 8; ++e) { B[e] = b0[e]; B[e + 8] = b1[e]; }
            acc[t] = __builtin_amdgcn_wmma_f32_16x16x32_f16(
                false, A, false, B, (short)0, acc[t], false, false);
        }
    }

    // combine the two half-row partial sums (lane L and L+16 own the same row M)
    lsum += __shfl_xor(lsum, 16, 32);
    float rinv = 1.f / fmaxf(lsum, 1e-30f);

#pragma unroll
    for (int r = 0; r < 8; ++r) {
        int m = r + 8 * hi;                 // C-layout row for this lane/vgpr
        float rv = __shfl(rinv, m, 32);     // lane m holds rowsum for row m
        int orow = r0 + m;
#pragma unroll
        for (int t = 0; t < 8; ++t)
            out[(size_t)orow * HIDc + t * 16 + lm] = acc[t][r] * rv;
    }
}

// ---------------------------------------------------------------
// Pooling: zero, segment-sum via atomics, final FC
// ---------------------------------------------------------------
__global__ __launch_bounds__(256) void k_zero(float* __restrict__ sums,
                                              float* __restrict__ cnt) {
    int i = blockIdx.x * blockDim.x + threadIdx.x;
    if (i < Gc * HIDc) sums[i] = 0.f;
    if (i < Gc) cnt[i] = 0.f;
}

__global__ __launch_bounds__(256) void k_pool(const float* __restrict__ h2,
                                              const int* __restrict__ batch,
                                              float* __restrict__ sums,
                                              float* __restrict__ cnt) {
    int idx = blockIdx.x * blockDim.x + threadIdx.x;
    if (idx >= Nn * HIDc) return;
    int i = idx >> 7, c = idx & 127;
    int g = batch[i];
    atomicAdd(&sums[(size_t)g * HIDc + c], h2[idx]);
    if (c == 0) atomicAdd(&cnt[g], 1.f);
}

__global__ __launch_bounds__(256) void k_out(const float* __restrict__ sums,
                                             const float* __restrict__ cnt,
                                             const float* __restrict__ fcw,
                                             const float* __restrict__ fcb,
                                             float* __restrict__ outv) {
    int g = blockIdx.x * blockDim.x + threadIdx.x;
    if (g >= Gc) return;
    float inv = 1.f / fmaxf(cnt[g], 1.f);
    float o = fcb[0];
#pragma unroll 16
    for (int c = 0; c < HIDc; ++c) o += sums[(size_t)g * HIDc + c] * inv * fcw[c];
    outv[g] = o;
}

// ---------------------------------------------------------------
extern "C" void kernel_launch(void* const* d_in, const int* in_sizes, int n_in,
                              void* d_out, int out_size, void* d_ws, size_t ws_size,
                              hipStream_t stream) {
    const float* x    = (const float*)d_in[0];
    const float* adj  = (const float*)d_in[1];
    const int*   bat  = (const int*)d_in[2];
    const float* W0   = (const float*)d_in[3];
    const float* a0s  = (const float*)d_in[4];
    const float* a0d  = (const float*)d_in[5];
    const float* W1   = (const float*)d_in[6];
    const float* a1s  = (const float*)d_in[7];
    const float* a1d  = (const float*)d_in[8];
    const float* W2   = (const float*)d_in[9];
    const float* a2s  = (const float*)d_in[10];
    const float* a2d  = (const float*)d_in[11];
    const float* fcw  = (const float*)d_in[12];
    const float* fcb  = (const float*)d_in[13];
    float* outv = (float*)d_out;

    // workspace carve-up (floats)
    float* w = (float*)d_ws;
    size_t o = 0;
    float* hA   = w + o; o += (size_t)Nn * HIDc;   // features h (per layer)
    float* hB   = w + o; o += (size_t)Nn * HIDc;   // attention output (per layer)
    float* sbuf = w + o; o += Nn;
    float* dbuf = w + o; o += Nn;
    float* Dm   = w + o; o += 4;                   // padded
    float* W1e  = w + o; o += HIDc * HIDc;
    float* W2e  = w + o; o += HIDc * HIDc;
    float* sums = w + o; o += Gc * HIDc;
    float* cnt  = w + o; o += Gc;
    o = (o + 3) & ~(size_t)3;                      // 16B-align the f16 region
    _Float16* ht = (_Float16*)(w + o);             // transposed f16 h [HID][N]

    // fold the tiled-head weights once
    k_fold<<<HIDc * HIDc / 128, 128, 0, stream>>>(W1, W1e);
    k_fold<<<HIDc * HIDc / 128, 128, 0, stream>>>(W2, W2e);

    const float* lin[3] = {x, hB, hB};
    const float* lW[3]  = {W0, W1e, W2e};
    const float* las[3] = {a0s, a1s, a2s};
    const float* lad[3] = {a0d, a1d, a2d};
    const int    lK[3]  = {ATOM, HIDc, HIDc};

    for (int l = 0; l < 3; ++l) {
        k_gemm<<<Nn / 16, 32, 0, stream>>>(lin[l], lW[l], hA, ht, lK[l], l > 0 ? 1 : 0);
        k_sd  <<<Nn / 256, 256, 0, stream>>>(hA, las[l], lad[l], sbuf, dbuf);
        k_maxd<<<1, 1024, 0, stream>>>(dbuf, Dm);
        k_attn<<<Nn / 128, 256, 0, stream>>>(adj, sbuf, dbuf, Dm, ht, hB);
    }

    k_zero<<<(Gc * HIDc + 255) / 256, 256, 0, stream>>>(sums, cnt);
    k_pool<<<(size_t)Nn * HIDc / 256, 256, 0, stream>>>(hB, bat, sums, cnt);
    k_out <<<1, 256, 0, stream>>>(sums, cnt, fcw, fcb, outv);
}
